// MloraMoE_65240553226580
// MI455X (gfx1250) — matrix operand order repaired and checked by reading, not compiled, not run on hardware
//
#include <hip/hip_runtime.h>
#include <hip/hip_bf16.h>

// ---------------------------------------------------------------------------
// MloraMoE for MI455X (gfx1250): bf16x3-split WMMA GEMMs, fp32 accumulate.
//   gate  = softmax(emb[dom] @ gW + gb)                       [B,8]
//   ts    = gate-scaled (x @ A_all)                           [B,128]
//   out   = relu(x@W + b + ts@Bflat + gate@lb)                [B,H]
// All matmuls use v_wmma_f32_16x16x32_bf16 with hi/lo splitting:
//   x @ W ~= x_hi@W_hi + x_hi@W_lo + x_lo@W_hi  (fp32-class accuracy)
// ---------------------------------------------------------------------------

typedef __attribute__((ext_vector_type(16))) __bf16 v16bf;
typedef __attribute__((ext_vector_type(8)))  __bf16 v8bf;
typedef __attribute__((ext_vector_type(8)))  float  v8f;

#define NB_TOTAL 16384
#define NEXP     8
#define LORA_RR  16
#define LORA_N   (NEXP * LORA_RR)   // 128

// ---- bf16 split helpers ----------------------------------------------------
__device__ __forceinline__ unsigned short f32_to_bf16_rn(float x) {
    unsigned u = __float_as_uint(x);
    unsigned r = u + 0x7FFFu + ((u >> 16) & 1u);   // round-to-nearest-even
    return (unsigned short)(r >> 16);
}

__device__ __forceinline__ void split2bf16(float x, unsigned short& hi, unsigned short& lo) {
    hi = f32_to_bf16_rn(x);
    float hif = __uint_as_float(((unsigned)hi) << 16);
    lo = f32_to_bf16_rn(x - hif);
}

// Build a v16bf fragment from two contiguous 8-element (16B-aligned) LDS runs.
__device__ __forceinline__ v16bf load_frag(const unsigned short* p0, const unsigned short* p1) {
    v8bf a = *(const v8bf*)p0;
    v8bf b = *(const v8bf*)p1;
    v16bf r;
#pragma unroll
    for (int i = 0; i < 8; ++i) { r[i] = a[i]; r[i + 8] = b[i]; }
    return r;
}

__device__ __forceinline__ v8f vzero8() {
    v8f z;
#pragma unroll
    for (int i = 0; i < 8; ++i) z[i] = 0.0f;
    return z;
}

// ---- gate kernel -----------------------------------------------------------
__global__ __launch_bounds__(256)
void gate_kernel(const int* __restrict__ dom,
                 const float* __restrict__ emb,   // [10,4]
                 const float* __restrict__ gW,    // [4,8]
                 const float* __restrict__ gb,    // [8]
                 float* __restrict__ gate)        // [B,8]
{
    int b = blockIdx.x * blockDim.x + threadIdx.x;
    if (b >= NB_TOTAL) return;
    int idx = dom[b];
    float d[4];
#pragma unroll
    for (int q = 0; q < 4; ++q) d[q] = emb[idx * 4 + q];
    float lg[NEXP];
    float mx = -1e30f;
#pragma unroll
    for (int e = 0; e < NEXP; ++e) {
        float a = gb[e];
#pragma unroll
        for (int q = 0; q < 4; ++q) a += d[q] * gW[q * NEXP + e];
        lg[e] = a;
        mx = a > mx ? a : mx;
    }
    float s = 0.0f;
#pragma unroll
    for (int e = 0; e < NEXP; ++e) { lg[e] = __expf(lg[e] - mx); s += lg[e]; }
    float inv = 1.0f / s;
#pragma unroll
    for (int e = 0; e < NEXP; ++e) gate[b * NEXP + e] = lg[e] * inv;
}

// ---- LoRA "t" kernel: ts[b, e*16+r] = gate[b,e] * (x @ A_e)[b,r] -----------
__global__ __launch_bounds__(256)
void lora_t_kernel(const float* __restrict__ X, int K,      // [B,K]
                   const float* __restrict__ A,             // [8,K,16]
                   const float* __restrict__ gate,          // [B,8]
                   float* __restrict__ TS)                  // [B,128]
{
    constexpr int BM = 64, KC = 64;
    __shared__ __align__(16) unsigned short sXhi[BM * KC];
    __shared__ __align__(16) unsigned short sXlo[BM * KC];
    __shared__ __align__(16) unsigned short sAhi[LORA_N * KC];  // transposed [n][k]
    __shared__ __align__(16) unsigned short sAlo[LORA_N * KC];
    __shared__ float sGate[BM * NEXP];

    const int tid  = threadIdx.x;
    const int wv   = tid >> 5, lane = tid & 31;
    const int rt   = wv >> 1;     // wave row-tile 0..3
    const int ch   = wv & 1;      // wave col-half 0..1
    const int row0 = blockIdx.y * BM;
    const int laneM = lane & 15, laneHi = lane >> 4;

    for (int i = tid; i < BM * NEXP; i += 256) sGate[i] = gate[row0 * NEXP + i];

    v8f acc[4];
#pragma unroll
    for (int j = 0; j < 4; ++j) acc[j] = vzero8();

    for (int kc = 0; kc < K; kc += KC) {
        __syncthreads();
        // X tile (f32 -> bf16 hi/lo)
        for (int f = tid; f < BM * (KC / 4); f += 256) {
            int r = f / (KC / 4);
            int c = (f % (KC / 4)) * 4;
            const float4 v = *(const float4*)&X[(size_t)(row0 + r) * K + kc + c];
            float vv[4] = {v.x, v.y, v.z, v.w};
#pragma unroll
            for (int q = 0; q < 4; ++q) {
                unsigned short h, l; split2bf16(vv[q], h, l);
                sXhi[r * KC + c + q] = h; sXlo[r * KC + c + q] = l;
            }
        }
        // A tile, transposed to [n][k] with n = e*16 + r
        for (int f = tid; f < LORA_N * KC; f += 256) {
            int c = f / KC;
            int k = f % KC;
            float v = A[(size_t)(c >> 4) * K * LORA_RR + (size_t)(kc + k) * LORA_RR + (c & 15)];
            unsigned short h, l; split2bf16(v, h, l);
            sAhi[c * KC + k] = h; sAlo[c * KC + k] = l;
        }
        __syncthreads();
#pragma unroll
        for (int ks = 0; ks < KC / 32; ++ks) {
            const int abase = (rt * 16 + laneM) * KC + ks * 32 + laneHi * 8;
            v16bf a_hi = load_frag(&sXhi[abase], &sXhi[abase + 16]);
            v16bf a_lo = load_frag(&sXlo[abase], &sXlo[abase + 16]);
#pragma unroll
            for (int j = 0; j < 4; ++j) {
                const int bbase = (ch * 64 + j * 16 + laneM) * KC + ks * 32 + laneHi * 8;
                v16bf b_hi = load_frag(&sAhi[bbase], &sAhi[bbase + 16]);
                v16bf b_lo = load_frag(&sAlo[bbase], &sAlo[bbase + 16]);
                acc[j] = __builtin_amdgcn_wmma_f32_16x16x32_bf16(false, a_hi, false, b_hi, (short)0, acc[j], false, false);
                acc[j] = __builtin_amdgcn_wmma_f32_16x16x32_bf16(false, a_hi, false, b_lo, (short)0, acc[j], false, false);
                acc[j] = __builtin_amdgcn_wmma_f32_16x16x32_bf16(false, a_lo, false, b_hi, (short)0, acc[j], false, false);
            }
        }
    }
    // epilogue: scale by gate and store ts (f32)
#pragma unroll
    for (int j = 0; j < 4; ++j) {
        const int n = ch * 64 + j * 16 + laneM;
        const int e = n >> 4;
#pragma unroll
        for (int v = 0; v < 8; ++v) {
            const int m = rt * 16 + laneHi * 8 + v;
            TS[(size_t)(row0 + m) * LORA_N + n] = acc[j][v] * sGate[m * NEXP + e];
        }
    }
}

// ---- main fused kernel: out = relu(x@W + b + ts@Bflat + gate@lb) -----------
__global__ __launch_bounds__(256)
void moe_main_kernel(const float* __restrict__ X, int K1,   // [B,K1]
                     const float* __restrict__ W,           // [K1,H]
                     const float* __restrict__ TS,          // [B,128]
                     const float* __restrict__ BF,          // [128,H]  (== B experts [8,16,H])
                     const float* __restrict__ bias,        // [H]
                     const float* __restrict__ lbias,       // [8,H]
                     const float* __restrict__ gate,        // [B,8]
                     float* __restrict__ out, int H)        // [B,H]
{
    constexpr int BM = 128, BN = 64, KC = 64;
    __shared__ __align__(16) unsigned short sXhi[BM * KC];
    __shared__ __align__(16) unsigned short sXlo[BM * KC];
    __shared__ __align__(16) unsigned short sWhi[BN * KC];  // transposed [n][k]
    __shared__ __align__(16) unsigned short sWlo[BN * KC];
    __shared__ float sGate[BM * NEXP];

    const int tid  = threadIdx.x;
    const int wv   = tid >> 5, lane = tid & 31;
    const int row0 = blockIdx.y * BM;
    const int col0 = blockIdx.x * BN;
    const int laneM = lane & 15, laneHi = lane >> 4;

    for (int i = tid; i < BM * NEXP; i += 256) sGate[i] = gate[row0 * NEXP + i];

    v8f acc[4];
#pragma unroll
    for (int j = 0; j < 4; ++j) acc[j] = vzero8();

    // pass 0: x @ W over K1 ; pass 1: ts @ Bflat over 128 — same accumulators
    for (int pass = 0; pass < 2; ++pass) {
        const float* Xp = pass ? TS : X;
        const float* Wp = pass ? BF : W;
        const int    Kp = pass ? LORA_N : K1;
        for (int kc = 0; kc < Kp; kc += KC) {
            __syncthreads();
            // X tile: BM x KC (f32 -> bf16 hi/lo)
            for (int f = tid; f < BM * (KC / 4); f += 256) {
                int r = f / (KC / 4);
                int c = (f % (KC / 4)) * 4;
                const float4 v = *(const float4*)&Xp[(size_t)(row0 + r) * Kp + kc + c];
                float vv[4] = {v.x, v.y, v.z, v.w};
#pragma unroll
                for (int q = 0; q < 4; ++q) {
                    unsigned short h, l; split2bf16(vv[q], h, l);
                    sXhi[r * KC + c + q] = h; sXlo[r * KC + c + q] = l;
                }
            }
            // W tile: KC x BN, stored transposed [n][k]
            for (int f = tid; f < KC * (BN / 4); f += 256) {
                int k = f / (BN / 4);
                int n = (f % (BN / 4)) * 4;
                const float4 v = *(const float4*)&Wp[(size_t)(kc + k) * H + col0 + n];
                float vv[4] = {v.x, v.y, v.z, v.w};
#pragma unroll
                for (int q = 0; q < 4; ++q) {
                    unsigned short h, l; split2bf16(vv[q], h, l);
                    sWhi[(n + q) * KC + k] = h; sWlo[(n + q) * KC + k] = l;
                }
            }
            __syncthreads();
#pragma unroll
            for (int ks = 0; ks < KC / 32; ++ks) {
                const int abase = (wv * 16 + laneM) * KC + ks * 32 + laneHi * 8;
                v16bf a_hi = load_frag(&sXhi[abase], &sXhi[abase + 16]);
                v16bf a_lo = load_frag(&sXlo[abase], &sXlo[abase + 16]);
#pragma unroll
                for (int j = 0; j < 4; ++j) {
                    const int bbase = (j * 16 + laneM) * KC + ks * 32 + laneHi * 8;
                    v16bf b_hi = load_frag(&sWhi[bbase], &sWhi[bbase + 16]);
                    v16bf b_lo = load_frag(&sWlo[bbase], &sWlo[bbase + 16]);
                    acc[j] = __builtin_amdgcn_wmma_f32_16x16x32_bf16(false, a_hi, false, b_hi, (short)0, acc[j], false, false);
                    acc[j] = __builtin_amdgcn_wmma_f32_16x16x32_bf16(false, a_hi, false, b_lo, (short)0, acc[j], false, false);
                    acc[j] = __builtin_amdgcn_wmma_f32_16x16x32_bf16(false, a_lo, false, b_hi, (short)0, acc[j], false, false);
                }
            }
        }
    }

    // epilogue: + bias + gate@lb, relu, store f32
#pragma unroll
    for (int j = 0; j < 4; ++j) {
        const int n = col0 + j * 16 + laneM;
        const float bv = bias[n];
        float lbv[NEXP];
#pragma unroll
        for (int e = 0; e < NEXP; ++e) lbv[e] = lbias[(size_t)e * H + n];
#pragma unroll
        for (int v = 0; v < 8; ++v) {
            const int m = wv * 16 + laneHi * 8 + v;
            float g = acc[j][v] + bv;
#pragma unroll
            for (int e = 0; e < NEXP; ++e) g += sGate[m * NEXP + e] * lbv[e];
            g = g > 0.0f ? g : 0.0f;
            out[(size_t)(row0 + m) * H + n] = g;
        }
    }
}

// ---------------------------------------------------------------------------
extern "C" void kernel_launch(void* const* d_in, const int* in_sizes, int n_in,
                              void* d_out, int out_size, void* d_ws, size_t ws_size,
                              hipStream_t stream)
{
    (void)in_sizes; (void)n_in; (void)out_size; (void)ws_size;

    const float* dnn = (const float*)d_in[0];
    const int*   dom = (const int*)d_in[1];
    const float* emb = (const float*)d_in[2];
    const float* gW  = (const float*)d_in[3];
    const float* gb  = (const float*)d_in[4];
    const float* Wl[3]  = {(const float*)d_in[5],  (const float*)d_in[10], (const float*)d_in[15]};
    const float* bl[3]  = {(const float*)d_in[6],  (const float*)d_in[11], (const float*)d_in[16]};
    const float* Al[3]  = {(const float*)d_in[7],  (const float*)d_in[12], (const float*)d_in[17]};
    const float* Bl[3]  = {(const float*)d_in[8],  (const float*)d_in[13], (const float*)d_in[18]};
    const float* lbl[3] = {(const float*)d_in[9],  (const float*)d_in[14], (const float*)d_in[19]};

    // workspace layout
    char*  ws   = (char*)d_ws;
    float* gate = (float*)ws;                                             // 16384*8*4   = 0.5 MB
    float* ts   = (float*)(ws + (size_t)NB_TOTAL * NEXP * 4);             // 16384*128*4 = 8 MB
    float* actA = (float*)(ws + (size_t)NB_TOTAL * NEXP * 4
                              + (size_t)NB_TOTAL * LORA_N * 4);           // 16384*1024*4 = 64 MB
    float* actB = (float*)((char*)actA + (size_t)NB_TOTAL * 1024 * 4);    // 16384*512*4  = 32 MB

    gate_kernel<<<NB_TOTAL / 256, 256, 0, stream>>>(dom, emb, gW, gb, gate);

    const int Pdims[3] = {1024, 1024, 512};
    const int Hdims[3] = {1024, 512, 256};
    const float* xin[3]  = {dnn, actA, actB};
    float*       xout[3] = {actA, actB, (float*)d_out};

    for (int li = 0; li < 3; ++li) {
        lora_t_kernel<<<dim3(1, NB_TOTAL / 64), 256, 0, stream>>>(
            xin[li], Pdims[li], Al[li], gate, ts);
        moe_main_kernel<<<dim3(Hdims[li] / 64, NB_TOTAL / 128), 256, 0, stream>>>(
            xin[li], Pdims[li], Wl[li], ts, Bl[li], bl[li], lbl[li], gate,
            xout[li], Hdims[li]);
    }
}
